// drugcell_nn_tta_64115271794767
// MI455X (gfx1250) — matrix operand order repaired and checked by reading, not compiled, not run on hardware
//
#include <hip/hip_runtime.h>
#include <hip/hip_bf16.h>
#include <math.h>

// ---------------- model dimensions ----------------
#define NGENE  3008
#define EDIM   8
#define HDIM   16
#define NDDIM  16
#define NTERMS 127
#define BB     8
#define LL     50
#define DD     128
#define NLAYER 8
#define INTERD 512
#define NHEADS 8
#define DHH    16
#define HFF    128
#define FINDIM (HDIM + DD)      // 144
#define BE     (BB * EDIM)      // 64 rows for the dg GEMM / tree samples
#define NCOLS  (NTERMS * NDDIM) // 2032
#define MTOK   (BB * LL)        // 400 tokens = exactly 25 row tiles

typedef __attribute__((ext_vector_type(16))) __bf16 v16bf;
typedef __attribute__((ext_vector_type(4)))  __bf16 v4bf;
typedef __attribute__((ext_vector_type(8)))  float  v8f;

// =====================================================================
// bf16 WMMA GEMM: C[M,N] = act(A[M,K] @ W[N,K]^T + bias[N])
// Compile-time N,K: static trip counts, immediate-offset addressing.
// REQUIRES: grid.y = M/16, N%16==0, K%32==0 (true at every call site).
// Block = 128 threads = 4 waves sharing one A tile; wave w owns N-tile
// blockIdx.x*64 + w*16. K loop software-pipelined with double-buffered
// LDS; global b128 loads for slab k+1 issue before the barrier while
// v_wmma consumes slab k.  ACT: 0 = none, 1 = relu
// =====================================================================
template <int ACT, int NN, int KK>
__global__ __launch_bounds__(128)
void gemm_bf16_wmma(const float* __restrict__ A, const float* __restrict__ W,
                    const float* __restrict__ bias, float* __restrict__ C)
{
    __shared__ __bf16 sA[2][16][40];        // 16 rows x 32 K, padded
    __shared__ __bf16 sB[2][4][16][40];     // per-wave B tiles
    const int tid   = threadIdx.x;
    const int lane  = tid & 31;
    const int wv    = tid >> 5;             // wave id 0..3
    const int mBase = blockIdx.y * 16;
    const int nBase = (blockIdx.x * 4 + wv) * 16;
    const bool nOK  = (nBase < NN);         // wave-uniform tail guard
    const int half  = lane >> 4;
    const int col   = lane & 15;

    // A tile: one float4 per thread (128 thr x 4 = 512 = 16x32)
    const int arow = tid >> 3;              // 0..15
    const int aq   = (tid & 7) * 4;         // 0,4,...,28
    const float* Aptr = A + (size_t)(mBase + arow) * KK + aq;
    // B tile: four float4 per lane (per wave)
    const float* Bptr[4];
    int brow[4], bq[4];
#pragma unroll
    for (int it = 0; it < 4; ++it) {
        const int idx = lane + 32 * it;     // 0..127
        brow[it] = idx >> 3;
        bq[it]   = (idx & 7) * 4;
        Bptr[it] = W + (size_t)(nBase + brow[it]) * KK + bq[it];
    }

    // ---- preload slab 0 into registers ----
    float4 ra = *(const float4*)(Aptr);
    float4 rb[4];
    if (nOK) {
#pragma unroll
        for (int it = 0; it < 4; ++it) rb[it] = *(const float4*)(Bptr[it]);
    }

    v8f acc = {};
    constexpr int nk = KK >> 5;             // static trip count
    for (int ks = 0; ks < nk; ++ks) {
        const int buf = ks & 1;
        // stage registers -> LDS as packed bf16 (8B ds stores)
        {
            v4bf pa;
            pa[0] = (__bf16)ra.x; pa[1] = (__bf16)ra.y;
            pa[2] = (__bf16)ra.z; pa[3] = (__bf16)ra.w;
            *(v4bf*)&sA[buf][arow][aq] = pa;
            if (nOK) {
#pragma unroll
                for (int it = 0; it < 4; ++it) {
                    v4bf pb;
                    pb[0] = (__bf16)rb[it].x; pb[1] = (__bf16)rb[it].y;
                    pb[2] = (__bf16)rb[it].z; pb[3] = (__bf16)rb[it].w;
                    *(v4bf*)&sB[buf][wv][brow[it]][bq[it]] = pb;
                }
            }
        }
        // issue next slab's global loads (consumed after next barrier)
        if (ks + 1 < nk) {
            const int koff = (ks + 1) * 32;
            ra = *(const float4*)(Aptr + koff);
            if (nOK) {
#pragma unroll
                for (int it = 0; it < 4; ++it)
                    rb[it] = *(const float4*)(Bptr[it] + koff);
            }
            if (ks + 2 < nk) {              // warm L2/WGP$ two slabs ahead
                __builtin_prefetch(Aptr + (ks + 2) * 32, 0, 3);
                if (nOK) __builtin_prefetch(Bptr[0] + (ks + 2) * 32, 0, 3);
            }
        }
        __syncthreads();
        if (nOK) {
            // gather per-lane operands into the wave32 WMMA layouts (ISA 7.12.2)
            v16bf a, b;
#pragma unroll
            for (int e = 0; e < 16; ++e) {
                const int v = e >> 1, w2 = e & 1;
                const int ka = (v < 4) ? (2 * v + w2 + 8 * half)
                                       : (8 + 2 * v + w2 + 8 * half);
                a[e] = sA[buf][col][ka];
                b[e] = sB[buf][wv][col][ka];
            }
            acc = __builtin_amdgcn_wmma_f32_16x16x32_bf16(
                      false, a, false, b, (short)0, acc, false, false);
        }
        // single barrier per iteration: next iteration writes buf^1, whose
        // readers all passed this barrier already.
    }
    if (nOK) {
        const int   n  = nBase + col;
        const float bv = bias[n];
        float* Crow = C + (size_t)(mBase + 8 * half) * NN + n;
#pragma unroll
        for (int r = 0; r < 8; ++r) {       // D layout: m = r + 8*half
            float o = acc[r] + bv;
            if (ACT == 1) o = fmaxf(o, 0.f);
            Crow[(size_t)r * NN] = o;       // constant row stride
        }
    }
}

// =====================================================================
// g[b,j] = cell_feature[b,j] * table[drug_idx[b], j]
// =====================================================================
__global__ __launch_bounds__(256)
void build_g_kernel(const float* __restrict__ cell, const int* __restrict__ drug_idx,
                    const int* __restrict__ table, float* __restrict__ g)
{
    int i = blockIdx.x * 256 + threadIdx.x;
    if (i < BB * NGENE) {
        int b = i / NGENE, j = i - b * NGENE;
        g[i] = cell[i] * (float)table[drug_idx[b] * NGENE + j];
    }
}

// =====================================================================
// Fused RBF + projection (never materialize the 290MB rbf matrix):
// geneA[(b*8+e), i] = sum_j exp(-0.1*(g_i-g_j)^2) * W_rbf[e,j] + b_rbf[e]
// =====================================================================
__global__ __launch_bounds__(256)
void rbf_gv_kernel(const float* __restrict__ g, const float* __restrict__ W_rbf,
                   const float* __restrict__ b_rbf, float* __restrict__ geneA)
{
    const int b   = blockIdx.y;
    const int i   = blockIdx.x * 256 + threadIdx.x;
    const int tid = threadIdx.x;
    __shared__ float sg[256];
    __shared__ float sw[EDIM][256];
    const float gi = (i < NGENE) ? g[b * NGENE + i] : 0.f;
    float acc[EDIM];
#pragma unroll
    for (int e = 0; e < EDIM; ++e) acc[e] = 0.f;

    for (int j0 = 0; j0 < NGENE; j0 += 256) {
        const int jj = j0 + tid;
        sg[tid] = (jj < NGENE) ? g[b * NGENE + jj] : 0.f;
#pragma unroll
        for (int e = 0; e < EDIM; ++e)
            sw[e][tid] = (jj < NGENE) ? W_rbf[e * NGENE + jj] : 0.f;
        __syncthreads();
        const int lim = (NGENE - j0 < 256) ? (NGENE - j0) : 256;
        for (int j = 0; j < lim; ++j) {
            const float d = gi - sg[j];
            const float w = __expf(-0.1f * d * d);
#pragma unroll
            for (int e = 0; e < EDIM; ++e) acc[e] = fmaf(w, sw[e][j], acc[e]);
        }
        __syncthreads();
    }
    if (i < NGENE) {
#pragma unroll
        for (int e = 0; e < EDIM; ++e)
            geneA[(size_t)(b * EDIM + e) * NGENE + i] = acc[e] + b_rbf[e];
    }
}

// =====================================================================
// Tree leaf level (64 nodes). 64 threads = (b,e) samples.
// allnn layout: [t][b][h][e]
// =====================================================================
__global__ __launch_bounds__(64)
void tree_leaf_kernel(const float* __restrict__ dgC, const float* __restrict__ W_leaf,
                      const float* __restrict__ b_leaf, const float* __restrict__ bn_g,
                      const float* __restrict__ bn_b, float* __restrict__ allnn)
{
    const int node = blockIdx.x;
    const int t    = 63 + node;
    const int be   = threadIdx.x;
    __shared__ float sh[HDIM][BE];
    __shared__ float smean[HDIM], sinv[HDIM];

    float xv[NDDIM];
#pragma unroll
    for (int i = 0; i < NDDIM; ++i) xv[i] = dgC[(size_t)be * NCOLS + t * NDDIM + i];
#pragma unroll
    for (int h = 0; h < HDIM; ++h) {
        float s = b_leaf[node * HDIM + h];
#pragma unroll
        for (int i = 0; i < NDDIM; ++i)
            s = fmaf(xv[i], W_leaf[(node * HDIM + h) * NDDIM + i], s);
        sh[h][be] = tanhf(s);
    }
    __syncthreads();
    if (be < HDIM) {
        float s = 0.f;
        for (int j = 0; j < BE; ++j) s += sh[be][j];
        const float m = s * (1.0f / BE);
        float v = 0.f;
        for (int j = 0; j < BE; ++j) { float d = sh[be][j] - m; v = fmaf(d, d, v); }
        smean[be] = m;
        sinv[be]  = rsqrtf(v * (1.0f / BE) + 1e-5f);
    }
    __syncthreads();
    const int b = be >> 3, e = be & 7;
#pragma unroll
    for (int h = 0; h < HDIM; ++h) {
        const float o = (sh[h][be] - smean[h]) * sinv[h] * bn_g[t * HDIM + h] + bn_b[t * HDIM + h];
        allnn[(((size_t)t * BB + b) * HDIM + h) * EDIM + e] = o;
    }
}

// =====================================================================
// Tree internal level. Input = [child_even H, child_odd H, dg ND] (48).
// =====================================================================
__global__ __launch_bounds__(64)
void tree_internal_kernel(int start, const float* __restrict__ dgC,
                          const float* __restrict__ W_int, const float* __restrict__ b_int,
                          const float* __restrict__ bn_g, const float* __restrict__ bn_b,
                          float* __restrict__ allnn)
{
    const int node = blockIdx.x;
    const int t    = start + node;
    const int cE   = 2 * t + 1, cO = 2 * t + 2;
    const int be   = threadIdx.x;
    const int b = be >> 3, e = be & 7;
    __shared__ float sh[HDIM][BE];
    __shared__ float smean[HDIM], sinv[HDIM];

    float xv[2 * HDIM + NDDIM];
#pragma unroll
    for (int h = 0; h < HDIM; ++h) {
        xv[h]        = allnn[(((size_t)cE * BB + b) * HDIM + h) * EDIM + e];
        xv[HDIM + h] = allnn[(((size_t)cO * BB + b) * HDIM + h) * EDIM + e];
    }
#pragma unroll
    for (int d = 0; d < NDDIM; ++d)
        xv[2 * HDIM + d] = dgC[(size_t)be * NCOLS + t * NDDIM + d];
#pragma unroll
    for (int h = 0; h < HDIM; ++h) {
        float s = b_int[t * HDIM + h];
#pragma unroll
        for (int i = 0; i < 2 * HDIM + NDDIM; ++i)
            s = fmaf(xv[i], W_int[((size_t)t * HDIM + h) * (2 * HDIM + NDDIM) + i], s);
        sh[h][be] = tanhf(s);
    }
    __syncthreads();
    if (be < HDIM) {
        float s = 0.f;
        for (int j = 0; j < BE; ++j) s += sh[be][j];
        const float m = s * (1.0f / BE);
        float v = 0.f;
        for (int j = 0; j < BE; ++j) { float d = sh[be][j] - m; v = fmaf(d, d, v); }
        smean[be] = m;
        sinv[be]  = rsqrtf(v * (1.0f / BE) + 1e-5f);
    }
    __syncthreads();
#pragma unroll
    for (int h = 0; h < HDIM; ++h) {
        const float o = (sh[h][be] - smean[h]) * sinv[h] * bn_g[t * HDIM + h] + bn_b[t * HDIM + h];
        allnn[(((size_t)t * BB + b) * HDIM + h) * EDIM + e] = o;
    }
}

// =====================================================================
// Aux heads: out[b][1+t]
// =====================================================================
__global__ __launch_bounds__(32)
void aux_kernel(const float* __restrict__ allnn, const float* __restrict__ Waux1,
                const float* __restrict__ baux1, const float* __restrict__ Waux2,
                const float* __restrict__ baux2, float* __restrict__ out)
{
    const int t = blockIdx.x;
    const int b = threadIdx.x;
    if (b < BB) {
        float s2 = baux2[t];
#pragma unroll
        for (int h = 0; h < HDIM; ++h) {
            float s = baux1[t];
#pragma unroll
            for (int e = 0; e < EDIM; ++e)
                s = fmaf(allnn[(((size_t)t * BB + b) * HDIM + h) * EDIM + e], Waux1[t * EDIM + e], s);
            s2 = fmaf(tanhf(s), Waux2[t * HDIM + h], s2);
        }
        out[b * 128 + 1 + t] = s2;
    }
}

// =====================================================================
// Embedding + LayerNorm
// =====================================================================
__global__ __launch_bounds__(DD)
void embed_ln_kernel(const int* __restrict__ drug_e, const float* __restrict__ tok_emb,
                     const float* __restrict__ pos_emb, const float* __restrict__ gln,
                     const float* __restrict__ bln, float* __restrict__ x)
{
    const int row = blockIdx.x;
    const int b = row / LL, l = row - b * LL;
    const int d = threadIdx.x;
    __shared__ float red[DD];
    const int tok = drug_e[b * LL + l];
    const float val = tok_emb[(size_t)tok * DD + d] + pos_emb[l * DD + d];
    red[d] = val; __syncthreads();
    for (int s = DD / 2; s > 0; s >>= 1) { if (d < s) red[d] += red[d + s]; __syncthreads(); }
    const float mean = red[0] * (1.0f / DD);
    __syncthreads();
    const float c = val - mean;
    red[d] = c * c; __syncthreads();
    for (int s = DD / 2; s > 0; s >>= 1) { if (d < s) red[d] += red[d + s]; __syncthreads(); }
    const float var = red[0] * (1.0f / DD);
    x[(size_t)row * DD + d] = c * rsqrtf(var + 1e-12f) * gln[d] + bln[d];
}

// =====================================================================
// x = LN(tmp + x)
// =====================================================================
__global__ __launch_bounds__(DD)
void ln_residual_kernel(const float* __restrict__ tmp, float* __restrict__ x,
                        const float* __restrict__ gln, const float* __restrict__ bln)
{
    const int row = blockIdx.x;
    const int d = threadIdx.x;
    __shared__ float red[DD];
    const float val = tmp[(size_t)row * DD + d] + x[(size_t)row * DD + d];
    red[d] = val; __syncthreads();
    for (int s = DD / 2; s > 0; s >>= 1) { if (d < s) red[d] += red[d + s]; __syncthreads(); }
    const float mean = red[0] * (1.0f / DD);
    __syncthreads();
    const float c = val - mean;
    red[d] = c * c; __syncthreads();
    for (int s = DD / 2; s > 0; s >>= 1) { if (d < s) red[d] += red[d + s]; __syncthreads(); }
    const float var = red[0] * (1.0f / DD);
    x[(size_t)row * DD + d] = c * rsqrtf(var + 1e-12f) * gln[d] + bln[d];
}

// =====================================================================
// Attention for one (batch, head)
// =====================================================================
__global__ __launch_bounds__(64)
void attention_kernel(const float* __restrict__ q, const float* __restrict__ k,
                      const float* __restrict__ v, const int* __restrict__ mask,
                      float* __restrict__ ctx)
{
    const int b = blockIdx.x, h = blockIdx.y;
    const int tid = threadIdx.x;
    __shared__ float sK[LL][DHH], sV[LL][DHH], sM[LL];
    for (int idx = tid; idx < LL * DHH; idx += 64) {
        const int r = idx >> 4, d = idx & 15;
        sK[r][d] = k[(size_t)(b * LL + r) * DD + h * DHH + d];
        sV[r][d] = v[(size_t)(b * LL + r) * DD + h * DHH + d];
    }
    for (int idx = tid; idx < LL; idx += 64)
        sM[idx] = (1.0f - (float)mask[b * LL + idx]) * -10000.0f;
    __syncthreads();
    if (tid < LL) {
        float qr[DHH];
#pragma unroll
        for (int d = 0; d < DHH; ++d) qr[d] = q[(size_t)(b * LL + tid) * DD + h * DHH + d];
        float s[LL];
        float mx = -3.0e38f;
#pragma unroll
        for (int kk = 0; kk < LL; ++kk) {
            float dot = 0.f;
#pragma unroll
            for (int d = 0; d < DHH; ++d) dot = fmaf(qr[d], sK[kk][d], dot);
            s[kk] = dot * 0.25f + sM[kk];
            mx = fmaxf(mx, s[kk]);
        }
        float denom = 0.f;
#pragma unroll
        for (int kk = 0; kk < LL; ++kk) { s[kk] = __expf(s[kk] - mx); denom += s[kk]; }
        const float inv = 1.0f / denom;
        float o[DHH];
#pragma unroll
        for (int d = 0; d < DHH; ++d) o[d] = 0.f;
#pragma unroll
        for (int kk = 0; kk < LL; ++kk) {
            const float w = s[kk] * inv;
#pragma unroll
            for (int d = 0; d < DHH; ++d) o[d] = fmaf(w, sV[kk][d], o[d]);
        }
#pragma unroll
        for (int d = 0; d < DHH; ++d)
            ctx[(size_t)(b * LL + tid) * DD + h * DHH + d] = o[d];
    }
}

// =====================================================================
// Final fusion head
// =====================================================================
__global__ __launch_bounds__(HFF)
void final_head_kernel(const float* __restrict__ allnn, const float* __restrict__ x,
                       const float* __restrict__ Wf, const float* __restrict__ bf,
                       const float* __restrict__ bn_f_g, const float* __restrict__ bn_f_b,
                       const float* __restrict__ Wfa, const float* __restrict__ bfa,
                       const float* __restrict__ Wfo, const float* __restrict__ bfo,
                       float* __restrict__ out)
{
    __shared__ float fin[BB][FINDIM];
    __shared__ float red[HFF][BB];
    const int tid = threadIdx.x;
    for (int idx = tid; idx < BB * FINDIM; idx += HFF) {
        const int b = idx / FINDIM, i = idx - b * FINDIM;
        float v;
        if (i < HDIM) {
            float s = 0.f;
#pragma unroll
            for (int e = 0; e < EDIM; ++e)
                s += allnn[(((size_t)0 * BB + b) * HDIM + i) * EDIM + e];
            v = s * (1.0f / EDIM);
        } else {
            v = x[(size_t)(b * LL) * DD + (i - HDIM)];
        }
        fin[b][i] = v;
    }
    __syncthreads();
    const int f = tid;
    float fh[BB];
#pragma unroll
    for (int b = 0; b < BB; ++b) {
        float s = bf[f];
        for (int i = 0; i < FINDIM; ++i) s = fmaf(fin[b][i], Wf[f * FINDIM + i], s);
        fh[b] = tanhf(s);
    }
    float m = 0.f;
#pragma unroll
    for (int b = 0; b < BB; ++b) m += fh[b];
    m *= (1.0f / BB);
    float var = 0.f;
#pragma unroll
    for (int b = 0; b < BB; ++b) { float d = fh[b] - m; var = fmaf(d, d, var); }
    var *= (1.0f / BB);
    const float sc = rsqrtf(var + 1e-5f) * bn_f_g[f];
#pragma unroll
    for (int b = 0; b < BB; ++b)
        red[f][b] = ((fh[b] - m) * sc + bn_f_b[f]) * Wfa[f];
    __syncthreads();
    if (tid < BB) {
        float s = bfa[0];
        for (int ff = 0; ff < HFF; ++ff) s += red[ff][tid];
        out[tid * 128 + 0] = tanhf(s) * Wfo[0] + bfo[0];
    }
}

// =====================================================================
// host launch
// =====================================================================
extern "C" void kernel_launch(void* const* d_in, const int* in_sizes, int n_in,
                              void* d_out, int out_size, void* d_ws, size_t ws_size,
                              hipStream_t stream)
{
    const float* cell_feature = (const float*)d_in[0];
    const int*   drug_e       = (const int*)  d_in[1];
    const int*   drug_mask    = (const int*)  d_in[2];
    const int*   drug_idx     = (const int*)  d_in[3];
    const int*   table        = (const int*)  d_in[4];
    const float* W_rbf   = (const float*)d_in[5];
    const float* b_rbf   = (const float*)d_in[6];
    const float* W_dg    = (const float*)d_in[7];
    const float* b_dg    = (const float*)d_in[8];
    const float* W_leaf  = (const float*)d_in[9];
    const float* b_leaf  = (const float*)d_in[10];
    const float* W_int   = (const float*)d_in[11];
    const float* b_int   = (const float*)d_in[12];
    const float* bn_g    = (const float*)d_in[13];
    const float* bn_b    = (const float*)d_in[14];
    const float* Waux1   = (const float*)d_in[15];
    const float* baux1   = (const float*)d_in[16];
    const float* Waux2   = (const float*)d_in[17];
    const float* baux2   = (const float*)d_in[18];
    const float* tok_emb = (const float*)d_in[19];
    const float* pos_emb = (const float*)d_in[20];
    const float* emb_ln_g= (const float*)d_in[21];
    const float* emb_ln_b= (const float*)d_in[22];
    const float* Wq = (const float*)d_in[23];
    const float* bq = (const float*)d_in[24];
    const float* Wk = (const float*)d_in[25];
    const float* bk = (const float*)d_in[26];
    const float* Wv = (const float*)d_in[27];
    const float* bv = (const float*)d_in[28];
    const float* Wo = (const float*)d_in[29];
    const float* bo = (const float*)d_in[30];
    const float* ln1_g = (const float*)d_in[31];
    const float* ln1_b = (const float*)d_in[32];
    const float* W1 = (const float*)d_in[33];
    const float* b1 = (const float*)d_in[34];
    const float* W2 = (const float*)d_in[35];
    const float* b2 = (const float*)d_in[36];
    const float* ln2_g = (const float*)d_in[37];
    const float* ln2_b = (const float*)d_in[38];
    const float* Wf  = (const float*)d_in[39];
    const float* bf_ = (const float*)d_in[40];
    const float* bn_f_g = (const float*)d_in[41];
    const float* bn_f_b = (const float*)d_in[42];
    const float* Wfa = (const float*)d_in[43];
    const float* bfa = (const float*)d_in[44];
    const float* Wfo = (const float*)d_in[45];
    const float* bfo = (const float*)d_in[46];
    float* out = (float*)d_out;

    // workspace layout (fp32, all offsets 16B aligned)
    float* ws    = (float*)d_ws;
    float* g     = ws;                                  // 24064
    float* geneA = g     + BB * NGENE;                  // 192512
    float* dgC   = geneA + BE * NGENE;                  // 130048
    float* allnn = dgC   + BE * NCOLS;                  // 130048
    float* x     = allnn + NTERMS * BB * HDIM * EDIM;   // 51200
    float* qb    = x     + MTOK * DD;
    float* kb    = qb    + MTOK * DD;
    float* vb    = kb    + MTOK * DD;
    float* ctxb  = vb    + MTOK * DD;
    float* tmp   = ctxb  + MTOK * DD;
    float* ffh   = tmp   + MTOK * DD;                   // 204800

    // ---- cell / gene branch ----
    build_g_kernel<<<(BB * NGENE + 255) / 256, 256, 0, stream>>>(cell_feature, drug_idx, table, g);
    rbf_gv_kernel<<<dim3((NGENE + 255) / 256, BB), 256, 0, stream>>>(g, W_rbf, b_rbf, geneA);
    // dg einsum as GEMM: (64 x 3008) @ (3008 x 2032); 127 N-tiles -> 32 blocks of 4 waves
    gemm_bf16_wmma<0, NCOLS, NGENE><<<dim3((NCOLS / 16 + 3) / 4, BE / 16), 128, 0, stream>>>(
        geneA, W_dg, b_dg, dgC);

    // ---- ontology tree ----
    tree_leaf_kernel<<<64, 64, 0, stream>>>(dgC, W_leaf, b_leaf, bn_g, bn_b, allnn);
    for (int lvl = 5; lvl >= 0; --lvl) {
        const int start = (1 << lvl) - 1;
        const int n     = 1 << lvl;
        tree_internal_kernel<<<n, 64, 0, stream>>>(start, dgC, W_int, b_int, bn_g, bn_b, allnn);
    }
    aux_kernel<<<NTERMS, 32, 0, stream>>>(allnn, Waux1, baux1, Waux2, baux2, out);

    // ---- drug transformer ----
    embed_ln_kernel<<<MTOK, DD, 0, stream>>>(drug_e, tok_emb, pos_emb, emb_ln_g, emb_ln_b, x);
    for (int i = 0; i < NLAYER; ++i) {
        const float* Wqi = Wq + (size_t)i * DD * DD;  const float* bqi = bq + i * DD;
        const float* Wki = Wk + (size_t)i * DD * DD;  const float* bki = bk + i * DD;
        const float* Wvi = Wv + (size_t)i * DD * DD;  const float* bvi = bv + i * DD;
        const float* Woi = Wo + (size_t)i * DD * DD;  const float* boi = bo + i * DD;
        const float* W1i = W1 + (size_t)i * INTERD * DD; const float* b1i = b1 + i * INTERD;
        const float* W2i = W2 + (size_t)i * DD * INTERD; const float* b2i = b2 + i * DD;

        gemm_bf16_wmma<0, DD, DD><<<dim3(DD / 64, MTOK / 16), 128, 0, stream>>>(x, Wqi, bqi, qb);
        gemm_bf16_wmma<0, DD, DD><<<dim3(DD / 64, MTOK / 16), 128, 0, stream>>>(x, Wki, bki, kb);
        gemm_bf16_wmma<0, DD, DD><<<dim3(DD / 64, MTOK / 16), 128, 0, stream>>>(x, Wvi, bvi, vb);
        attention_kernel<<<dim3(BB, NHEADS), 64, 0, stream>>>(qb, kb, vb, drug_mask, ctxb);
        gemm_bf16_wmma<0, DD, DD><<<dim3(DD / 64, MTOK / 16), 128, 0, stream>>>(ctxb, Woi, boi, tmp);
        ln_residual_kernel<<<MTOK, DD, 0, stream>>>(tmp, x, ln1_g + i * DD, ln1_b + i * DD);
        gemm_bf16_wmma<1, INTERD, DD><<<dim3(INTERD / 64, MTOK / 16), 128, 0, stream>>>(x, W1i, b1i, ffh);
        gemm_bf16_wmma<0, DD, INTERD><<<dim3(DD / 64, MTOK / 16), 128, 0, stream>>>(ffh, W2i, b2i, tmp);
        ln_residual_kernel<<<MTOK, DD, 0, stream>>>(tmp, x, ln2_g + i * DD, ln2_b + i * DD);
    }

    // ---- fusion head ----
    final_head_kernel<<<1, HFF, 0, stream>>>(allnn, x, Wf, bf_, bn_f_g, bn_f_b,
                                             Wfa, bfa, Wfo, bfo, out);

    (void)in_sizes; (void)n_in; (void)out_size; (void)ws_size;
}
